// GNN_38740605010070
// MI455X (gfx1250) — matrix operand order repaired and checked by reading, not compiled, not run on hardware
//
#include <hip/hip_runtime.h>
#include <hip/hip_bf16.h>
#include <stdint.h>

// ---------------------------------------------------------------------------
// GAT layer for MI455X (gfx1250): split-bf16 WMMA GEMM (with fused attention
// logit reduction) + edge-parallel segment softmax + atomic scatter
// aggregation + fused ReLU/BatchNorm.
// ---------------------------------------------------------------------------

#define D_IN   256
#define D_HID  128

typedef __attribute__((ext_vector_type(16))) __bf16 v16bf;
typedef __attribute__((ext_vector_type(8)))  float  v8f;

union Frag { v16bf v; unsigned u[8]; };

// ---- bf16 split helpers (round-to-nearest-even) ---------------------------
__device__ __forceinline__ unsigned bf16_bits(float f) {
    unsigned b = __float_as_uint(f);
    unsigned r = b + 0x7FFFu + ((b >> 16) & 1u);
    return r >> 16;
}
__device__ __forceinline__ float bf16_val(float f) {
    return __uint_as_float(bf16_bits(f) << 16);
}
__device__ __forceinline__ unsigned pack_hi(float f0, float f1) {
    return bf16_bits(f0) | (bf16_bits(f1) << 16);
}
__device__ __forceinline__ unsigned pack_lo(float f0, float f1) {
    return bf16_bits(f0 - bf16_val(f0)) | (bf16_bits(f1 - bf16_val(f1)) << 16);
}

// ---- order-preserving float <-> uint for atomicMax ------------------------
__device__ __forceinline__ unsigned enc_f(float x) {
    unsigned b = __float_as_uint(x);
    return (b & 0x80000000u) ? ~b : (b | 0x80000000u);
}
__device__ __forceinline__ float dec_f(unsigned u) {
    unsigned b = (u & 0x80000000u) ? (u & 0x7FFFFFFFu) : ~u;
    return __uint_as_float(b);
}

// ---------------------------------------------------------------------------
// Pack W [256,128] f32 into (k,k+1) bf16 pairs: Wph/Wpl are [128][128] dwords.
// ---------------------------------------------------------------------------
__global__ void gat_prep_w(const float* __restrict__ W,
                           unsigned* __restrict__ Wph,
                           unsigned* __restrict__ Wpl) {
    int idx = blockIdx.x * blockDim.x + threadIdx.x;
    if (idx >= (D_IN / 2) * D_HID) return;
    int kp = idx >> 7;            // k-pair index 0..127
    int n  = idx & 127;           // column
    float w0 = W[(2 * kp + 0) * D_HID + n];
    float w1 = W[(2 * kp + 1) * D_HID + n];
    Wph[idx] = pack_hi(w0, w1);
    Wpl[idx] = pack_lo(w0, w1);
}

// ---------------------------------------------------------------------------
// h = x @ W via v_wmma_f32_16x16x32_bf16, split-bf16 (3 products) for ~fp32
// accuracy. Block = 256 threads = 8 waves; each wave owns one 16x16 C tile.
// Fused epilogue also produces a_s[n] = h[n,:].att_src and a_d likewise,
// reduced deterministically via lane shuffles + LDS (no atomics).
// ---------------------------------------------------------------------------
__global__ void gat_gemm_wmma(const float* __restrict__ x,
                              const unsigned* __restrict__ Wph,
                              const unsigned* __restrict__ Wpl,
                              const float* __restrict__ att_s,
                              const float* __restrict__ att_d,
                              float* __restrict__ h,
                              float* __restrict__ a_s,
                              float* __restrict__ a_d, int N) {
    __shared__ float red_s[8][16];
    __shared__ float red_d[8][16];

    const int lane = threadIdx.x & 31;
    const int wave = threadIdx.x >> 5;
    const int hf   = lane >> 4;          // lane half (K-split)
    const int l15  = lane & 15;
    const int rowbase = blockIdx.x * 16;
    int arow = rowbase + l15;            // A row this lane feeds
    if (arow >= N) arow = N - 1;         // clamp (guarded on store)
    const int ncol = wave * 16 + l15;    // B/C column this lane owns

    v8f acc = {0.f, 0.f, 0.f, 0.f, 0.f, 0.f, 0.f, 0.f};

    for (int kb = 0; kb < D_IN / 32; ++kb) {
        // ---- A fragment: 16x32 bf16. Lane half hf: VGPR v=0..3 -> K =
        // kb*32 + hf*8 + 2v{,+1}; v=4..7 -> +16. Pairs contiguous in x.
        const float* xr = x + (size_t)arow * D_IN + kb * 32 + hf * 8;
        float4 f0 = *(const float4*)(xr);
        float4 f1 = *(const float4*)(xr + 4);
        float4 f2 = *(const float4*)(xr + 16);
        float4 f3 = *(const float4*)(xr + 20);
        Frag ah, al;
        ah.u[0] = pack_hi(f0.x, f0.y); ah.u[1] = pack_hi(f0.z, f0.w);
        ah.u[2] = pack_hi(f1.x, f1.y); ah.u[3] = pack_hi(f1.z, f1.w);
        ah.u[4] = pack_hi(f2.x, f2.y); ah.u[5] = pack_hi(f2.z, f2.w);
        ah.u[6] = pack_hi(f3.x, f3.y); ah.u[7] = pack_hi(f3.z, f3.w);
        al.u[0] = pack_lo(f0.x, f0.y); al.u[1] = pack_lo(f0.z, f0.w);
        al.u[2] = pack_lo(f1.x, f1.y); al.u[3] = pack_lo(f1.z, f1.w);
        al.u[4] = pack_lo(f2.x, f2.y); al.u[5] = pack_lo(f2.z, f2.w);
        al.u[6] = pack_lo(f3.x, f3.y); al.u[7] = pack_lo(f3.z, f3.w);

        // ---- B fragment: 32x16 bf16. Lane half hf: VGPR v=0..3 -> K =
        // hf*8 + 2v{,+1}; v=4..7 -> +16 (matches A).
        Frag bh, bl;
        #pragma unroll
        for (int v = 0; v < 8; ++v) {
            int kp = kb * 16 + (v >> 2) * 8 + hf * 4 + (v & 3); // k-pair index in W
            bh.u[v] = Wph[kp * D_HID + ncol];
            bl.u[v] = Wpl[kp * D_HID + ncol];
        }

        // h += xh*Wh + xh*Wl + xl*Wh  (xl*Wl negligible)
        acc = __builtin_amdgcn_wmma_f32_16x16x32_bf16(
                  false, ah.v, false, bh.v, (short)0, acc, false, false);
        acc = __builtin_amdgcn_wmma_f32_16x16x32_bf16(
                  false, ah.v, false, bl.v, (short)0, acc, false, false);
        acc = __builtin_amdgcn_wmma_f32_16x16x32_bf16(
                  false, al.v, false, bh.v, (short)0, acc, false, false);
    }

    // ---- store C: VGPR i -> row (i + hf*8), col = l15 within tile.
    // Uniform bounds branch (full tiles take the unguarded fast path).
    float* hp = h + (size_t)(rowbase + hf * 8) * D_HID + wave * 16 + l15;
    if (rowbase + 16 <= N) {
        #pragma unroll
        for (int i = 0; i < 8; ++i)
            hp[(size_t)i * D_HID] = acc[i];
    } else {
        #pragma unroll
        for (int i = 0; i < 8; ++i)
            if (rowbase + hf * 8 + i < N) hp[(size_t)i * D_HID] = acc[i];
    }

    // ---- fused attention logits: per-row dot with att_src / att_dst.
    // Reduce across the 16 columns of this wave's tile (xor offsets < 16
    // stay within a lane half), stage per-wave partials in LDS, combine.
    float ws_s = att_s[ncol];
    float ws_d = att_d[ncol];
    #pragma unroll
    for (int i = 0; i < 8; ++i) {
        float ps = acc[i] * ws_s;
        float pd = acc[i] * ws_d;
        #pragma unroll
        for (int off = 8; off >= 1; off >>= 1) {
            ps += __shfl_xor(ps, off, 32);
            pd += __shfl_xor(pd, off, 32);
        }
        if (l15 == 0) {
            red_s[wave][i + hf * 8] = ps;
            red_d[wave][i + hf * 8] = pd;
        }
    }
    __syncthreads();
    if (threadIdx.x < 16) {
        int row = rowbase + threadIdx.x;
        if (row < N) {
            float ss = 0.f, sd = 0.f;
            #pragma unroll
            for (int w = 0; w < 8; ++w) {
                ss += red_s[w][threadIdx.x];
                sd += red_d[w][threadIdx.x];
            }
            a_s[row] = ss;
            a_d[row] = sd;
        }
    }
}

// ---------------------------------------------------------------------------
// Edge helpers: edges 0..E-1 from edge_index (int64, [2,E] row-major),
// edges E..E+N-1 are self-loops.
// ---------------------------------------------------------------------------
__device__ __forceinline__ void edge_sd(const long long* __restrict__ ei,
                                        long long E, long long e,
                                        int& s, int& d) {
    if (e < E) { s = (int)ei[e]; d = (int)ei[E + e]; }
    else       { s = (int)(e - E); d = s; }
}
__device__ __forceinline__ float leaky(float v) {
    return v > 0.f ? v : 0.2f * v;
}

// Pass 1: per-edge leaky logit -> elog[e] (streamed), segment max via
// order-preserving uint atomicMax.
__global__ void gat_edge_max(const long long* __restrict__ ei,
                             const float* __restrict__ a_s,
                             const float* __restrict__ a_d,
                             float* __restrict__ elog,
                             unsigned* __restrict__ menc,
                             long long E, int N) {
    long long e = (long long)blockIdx.x * blockDim.x + threadIdx.x;
    if (e >= E + N) return;
    int s, d; edge_sd(ei, E, e, s, d);
    float v = leaky(a_s[s] + a_d[d]);
    elog[e] = v;
    atomicMax(&menc[d], enc_f(v));
}

// Pass 2: exp(e - m[dst]) and segment sum.
__global__ void gat_edge_sum(const long long* __restrict__ ei,
                             const float* __restrict__ elog,
                             const unsigned* __restrict__ menc,
                             float* __restrict__ ssum,
                             long long E, int N) {
    long long e = (long long)blockIdx.x * blockDim.x + threadIdx.x;
    if (e >= E + N) return;
    int s, d; edge_sd(ei, E, e, s, d);
    float w = __expf(elog[e] - dec_f(menc[d]));
    atomicAdd(&ssum[d], w);
}

// Pass 3: one wave32 per edge: out[dst,:] += alpha * h[src,:] (128 cols).
__global__ void gat_aggregate(const long long* __restrict__ ei,
                              const float* __restrict__ elog,
                              const unsigned* __restrict__ menc,
                              const float* __restrict__ ssum,
                              const float* __restrict__ h,
                              float* __restrict__ out,
                              long long E, int N) {
    long long wid = ((long long)blockIdx.x * blockDim.x + threadIdx.x) >> 5;
    int lane = threadIdx.x & 31;
    if (wid >= E + N) return;
    int s, d; edge_sd(ei, E, wid, s, d);
    float alpha = __expf(elog[wid] - dec_f(menc[d])) / (ssum[d] + 1e-16f);
    const float* hr = h + (size_t)s * D_HID;
    float* orow = out + (size_t)d * D_HID;
    #pragma unroll
    for (int j = 0; j < 4; ++j) {
        int c = lane + j * 32;
        atomicAdd(&orow[c], alpha * hr[c]);
    }
}

// ---------------------------------------------------------------------------
// z = relu(out + b); write back; accumulate per-column sum / sum-of-squares
// via LDS (ds_add_f32) partials + one global atomic per column per block.
// ---------------------------------------------------------------------------
__global__ void gat_bias_relu_stats(float* __restrict__ out,
                                    const float* __restrict__ bias,
                                    float* __restrict__ cs,
                                    float* __restrict__ cs2,
                                    long long total) {
    __shared__ float ls[D_HID];
    __shared__ float ls2[D_HID];
    if (threadIdx.x < D_HID) { ls[threadIdx.x] = 0.f; ls2[threadIdx.x] = 0.f; }
    __syncthreads();
    long long stride = (long long)gridDim.x * blockDim.x;
    for (long long i = (long long)blockIdx.x * blockDim.x + threadIdx.x;
         i < total; i += stride) {
        int c = (int)(i & (D_HID - 1));
        float z = fmaxf(out[i] + bias[c], 0.f);
        out[i] = z;
        atomicAdd(&ls[c], z);
        atomicAdd(&ls2[c], z * z);
    }
    __syncthreads();
    if (threadIdx.x < D_HID) {
        atomicAdd(&cs[threadIdx.x], ls[threadIdx.x]);
        atomicAdd(&cs2[threadIdx.x], ls2[threadIdx.x]);
    }
}

__global__ void gat_bn_finalize(const float* __restrict__ cs,
                                const float* __restrict__ cs2,
                                const float* __restrict__ gamma,
                                const float* __restrict__ beta,
                                float* __restrict__ scale,
                                float* __restrict__ shift, int N) {
    int c = threadIdx.x;
    if (c < D_HID) {
        float inv = 1.f / (float)N;
        float mu  = cs[c] * inv;
        float var = cs2[c] * inv - mu * mu;   // biased (jnp.var default)
        float sc  = rsqrtf(var + 1e-5f) * gamma[c];
        scale[c]  = sc;
        shift[c]  = beta[c] - mu * sc;
    }
}

__global__ void gat_bn_apply(float* __restrict__ out,
                             const float* __restrict__ scale,
                             const float* __restrict__ shift,
                             long long total) {
    long long stride = (long long)gridDim.x * blockDim.x;
    for (long long i = (long long)blockIdx.x * blockDim.x + threadIdx.x;
         i < total; i += stride) {
        int c = (int)(i & (D_HID - 1));
        out[i] = out[i] * scale[c] + shift[c];
    }
}

// ---------------------------------------------------------------------------
// Launch
// ---------------------------------------------------------------------------
extern "C" void kernel_launch(void* const* d_in, const int* in_sizes, int n_in,
                              void* d_out, int out_size, void* d_ws, size_t ws_size,
                              hipStream_t stream) {
    const float*      x     = (const float*)d_in[0];
    const long long*  ei    = (const long long*)d_in[1];   // int64 [2,E]
    const float*      W     = (const float*)d_in[2];
    const float*      b     = (const float*)d_in[3];
    const float*      att_s = (const float*)d_in[4];
    const float*      att_d = (const float*)d_in[5];
    const float*      gamma = (const float*)d_in[6];
    const float*      beta  = (const float*)d_in[7];
    float*            out   = (float*)d_out;

    const int       N = in_sizes[0] / D_IN;          // 100000
    const long long E = in_sizes[1] / 2;             // 1600000
    const long long T = E + N;                       // edges incl. self-loops
    const long long total = (long long)N * D_HID;

    // ---- carve workspace (256B aligned slabs) ----
    char* p = (char*)d_ws;
    auto carve = [&](size_t bytes) -> char* {
        char* r = p;
        p += (bytes + 255) & ~(size_t)255;
        return r;
    };
    float*    h     = (float*)   carve((size_t)N * D_HID * sizeof(float));
    float*    a_s   = (float*)   carve((size_t)N * sizeof(float));
    float*    a_d   = (float*)   carve((size_t)N * sizeof(float));
    unsigned* menc  = (unsigned*)carve((size_t)N * sizeof(unsigned));
    float*    ssum  = (float*)   carve((size_t)N * sizeof(float));
    float*    elog  = (float*)   carve((size_t)T * sizeof(float));
    unsigned* Wph   = (unsigned*)carve((size_t)(D_IN / 2) * D_HID * sizeof(unsigned));
    unsigned* Wpl   = (unsigned*)carve((size_t)(D_IN / 2) * D_HID * sizeof(unsigned));
    float*    cs    = (float*)   carve(2 * D_HID * sizeof(float)); // cs | cs2
    float*    cs2   = cs + D_HID;
    float*    scale = (float*)   carve(2 * D_HID * sizeof(float)); // scale | shift
    float*    shift = scale + D_HID;
    (void)ws_size; (void)n_in; (void)out_size;

    // ---- zero-init accumulators (graph-capture-safe async memsets) ----
    hipMemsetAsync(out,  0, (size_t)total * sizeof(float), stream);
    hipMemsetAsync(menc, 0, (size_t)N * sizeof(unsigned), stream);  // enc < all reals
    hipMemsetAsync(ssum, 0, (size_t)N * sizeof(float), stream);
    hipMemsetAsync(cs,   0, 2 * D_HID * sizeof(float), stream);

    // ---- 1) pack W into split-bf16 pair form ----
    gat_prep_w<<<((D_IN / 2) * D_HID + 255) / 256, 256, 0, stream>>>(W, Wph, Wpl);

    // ---- 2) WMMA GEMM: h = x @ W, fused a_s / a_d reduction ----
    gat_gemm_wmma<<<(N + 15) / 16, 256, 0, stream>>>(
        x, Wph, Wpl, att_s, att_d, h, a_s, a_d, N);

    // ---- 3) segment max / 4) segment sum / 5) weighted scatter ----
    gat_edge_max<<<(int)((T + 255) / 256), 256, 0, stream>>>(
        ei, a_s, a_d, elog, menc, E, N);
    gat_edge_sum<<<(int)((T + 255) / 256), 256, 0, stream>>>(
        ei, elog, menc, ssum, E, N);
    gat_aggregate<<<(int)((T * 32 + 255) / 256), 256, 0, stream>>>(
        ei, elog, menc, ssum, h, out, E, N);

    // ---- 6) bias + ReLU + column stats, 7) BN finalize, 8) BN apply ----
    gat_bias_relu_stats<<<2048, 256, 0, stream>>>(out, b, cs, cs2, total);
    gat_bn_finalize<<<1, D_HID, 0, stream>>>(cs, cs2, gamma, beta, scale, shift, N);
    gat_bn_apply<<<4096, 256, 0, stream>>>(out, scale, shift, total);
}
